// CA_54889682043704
// MI455X (gfx1250) — matrix operand order repaired and checked by reading, not compiled, not run on hardware
//
#include <hip/hip_runtime.h>
#include <math.h>

#define B_SZ   4
#define C_DIM  512
#define N_DIM  4096
#define M_DIM  4096
#define DA_DIM 128
#define BN_EPS 1e-5f

typedef __attribute__((ext_vector_type(16))) __bf16          bf16x16;
typedef __attribute__((ext_vector_type(8)))  float           v8f;
typedef __attribute__((ext_vector_type(8)))  unsigned int    u32x8;
typedef __attribute__((ext_vector_type(4)))  unsigned int    u32x4;
typedef __attribute__((ext_vector_type(8)))  unsigned short  u16x8;
typedef __attribute__((ext_vector_type(16))) unsigned short  u16x16;

__device__ __forceinline__ unsigned short f2bf(float f) {
    unsigned u = __float_as_uint(f);
    unsigned r = u + 0x7FFFu + ((u >> 16) & 1u);   // round-to-nearest-even
    return (unsigned short)(r >> 16);
}
// SWAPX16 lane exchange (xor=0x10, or=0, and=0x1f) -> one ds_swizzle
__device__ __forceinline__ float swz16f(float x) {
    return __int_as_float(__builtin_amdgcn_ds_swizzle(__float_as_int(x), 0x401F));
}
__device__ __forceinline__ unsigned swz16u(unsigned x) {
    return (unsigned)__builtin_amdgcn_ds_swizzle((int)x, 0x401F);
}
// A-operand 16-bit fragment from two contiguous 8-element chunks
__device__ __forceinline__ bf16x16 combine8(u16x8 lo, u16x8 hi) {
    u16x16 v;
#pragma unroll
    for (int i = 0; i < 8; ++i) { v[i] = lo[i]; v[i + 8] = hi[i]; }
    return __builtin_bit_cast(bf16x16, v);
}
__device__ __forceinline__ bf16x16 load_afrag(const unsigned short* p) {
    return combine8(*(const u16x8*)p, *(const u16x8*)(p + 8));
}
__device__ __forceinline__ bf16x16 load_bfrag(const unsigned short* p) {
    return __builtin_bit_cast(bf16x16, *(const u16x16*)p);
}

// ---------------------------------------------------------------------------
// Transpose + fp32->bf16: Out[b,p,c] = bf16(In[b,c,p]).  Block 32x8, LDS tile.
// ---------------------------------------------------------------------------
__global__ __launch_bounds__(256) void transpose_cvt_kernel(
    const float* __restrict__ In, unsigned short* __restrict__ Out, int P)
{
    __shared__ unsigned short t[32][33];
    const int px = threadIdx.x, py = threadIdx.y;
    const int p0 = blockIdx.x * 32, c0 = blockIdx.y * 32, b = blockIdx.z;
#pragma unroll
    for (int i = 0; i < 4; ++i) {
        int c = c0 + py + i * 8;
        t[py + i * 8][px] = f2bf(In[((size_t)b * C_DIM + c) * P + p0 + px]);
    }
    __syncthreads();
#pragma unroll
    for (int i = 0; i < 4; ++i) {
        int p = p0 + py + i * 8;
        Out[((size_t)b * P + p) * C_DIM + c0 + px] = t[px][py + i * 8];
    }
}

// flat fp32 -> bf16 weight convert
__global__ __launch_bounds__(256) void cvt_kernel(
    const float* __restrict__ In, unsigned short* __restrict__ Out, int n)
{
    int i = blockIdx.x * 256 + threadIdx.x;
    if (i < n) Out[i] = f2bf(In[i]);
}

// ---------------------------------------------------------------------------
// Shared WMMA core, ping-pong software pipeline over K=512 (16 chunks of 32):
// acc[j] += Asrc[arow,:]·Bsrc[brow0+j*16+lrow,:]   (both row-major, rowlen 512)
// ---------------------------------------------------------------------------
__device__ __forceinline__ void load_group(const unsigned short* ap,
                                           const unsigned short* bp, int c,
                                           bf16x16& af, bf16x16 bf[4])
{
    af = load_afrag(ap + c);
#pragma unroll
    for (int j = 0; j < 4; ++j)
        bf[j] = load_bfrag(bp + c + (size_t)j * 16 * C_DIM);
}
__device__ __forceinline__ void wmma_group(bf16x16 af, const bf16x16 bf[4],
                                           v8f acc[4])
{
#pragma unroll
    for (int j = 0; j < 4; ++j)
        acc[j] = __builtin_amdgcn_wmma_f32_16x16x32_bf16(
                     false, af, false, bf[j], (short)0, acc[j], false, false);
}
__device__ __forceinline__ void wmma_k512(const unsigned short* aRow,
                                          const unsigned short* bRow,
                                          int lhi, v8f acc[4])
{
    const unsigned short* ap = aRow + 16 * lhi;
    const unsigned short* bp = bRow + 16 * lhi;

    bf16x16 a0, a1, b0[4], b1[4];
    load_group(ap, bp, 0, a0, b0);
    for (int i = 0; i < 7; ++i) {          // chunks 0..13 (ping-pong, no movs)
        int c = i * 64;
        load_group(ap, bp, c + 32, a1, b1);
        wmma_group(a0, b0, acc);
        load_group(ap, bp, c + 64, a0, b0);
        wmma_group(a1, b1, acc);
    }
    load_group(ap, bp, C_DIM - 32, a1, b1); // chunk 15
    wmma_group(a0, b0, acc);                // chunk 14
    wmma_group(a1, b1, acc);                // chunk 15
}

// ---- q/k GEMM: Out[b,p,o] = scale * (W[o,:]·InT[b,p,:]).  Wave: 16p x 64o.
__global__ __launch_bounds__(256) void gemm_po_kernel(
    const unsigned short* __restrict__ InT,  // [B, P, 512]
    const unsigned short* __restrict__ Wb,   // [O, 512]
    unsigned short* __restrict__ Out,        // [B, P, O]
    int P, int O, float scale)
{
    const int lane = threadIdx.x & 31, wave = threadIdx.x >> 5;
    const int lrow = lane & 15, lhi = lane >> 4;
    const int b = blockIdx.z;
    const int prow0 = blockIdx.y * 128 + wave * 16;   // A side
    const int orow0 = blockIdx.x * 64;                // B side
    v8f acc[4] = {};
    wmma_k512(InT + ((size_t)b * P + prow0 + lrow) * C_DIM,
              Wb + (size_t)(orow0 + lrow) * C_DIM, lhi, acc);
#pragma unroll
    for (int j = 0; j < 4; ++j)
#pragma unroll
        for (int v = 0; v < 8; ++v) {
            int p = prow0 + lhi * 8 + v;
            int o = orow0 + j * 16 + lrow;
            Out[((size_t)b * P + p) * O + o] = f2bf(acc[j][v] * scale);
        }
}

// ---- v GEMM: Out[b,o,p] = W[o,:]·InT[b,p,:] + bias[o].  Wave: 16o x 64p.
__global__ __launch_bounds__(256) void gemm_op_kernel(
    const unsigned short* __restrict__ Wb,   // [O, 512]
    const unsigned short* __restrict__ InT,  // [B, P, 512]
    const float* __restrict__ bias,
    unsigned short* __restrict__ Out,        // [B, O, P]
    int O, int P)
{
    const int lane = threadIdx.x & 31, wave = threadIdx.x >> 5;
    const int lrow = lane & 15, lhi = lane >> 4;
    const int b = blockIdx.z;
    const int orow0 = blockIdx.y * 16;                    // A side
    const int prow0 = blockIdx.x * 512 + wave * 64;       // B side
    v8f acc[4] = {};
    wmma_k512(Wb + (size_t)(orow0 + lrow) * C_DIM,
              InT + ((size_t)b * P + prow0 + lrow) * C_DIM, lhi, acc);
#pragma unroll
    for (int j = 0; j < 4; ++j)
#pragma unroll
        for (int v = 0; v < 8; ++v) {
            int o = orow0 + lhi * 8 + v;
            int p = prow0 + j * 16 + lrow;
            Out[((size_t)b * O + o) * P + p] = f2bf(acc[j][v] + bias[o]);
        }
}

// ---- final GEMM + BN(eval) + ReLU: out[b,n,o] fp32.  Wave: 16n x 64o.
__global__ __launch_bounds__(256) void gemm_final_kernel(
    const unsigned short* __restrict__ XsT,  // [B, N, 512]
    const unsigned short* __restrict__ Wtb,  // [512, 512]
    const float* __restrict__ bt, const float* __restrict__ gamma,
    const float* __restrict__ beta, const float* __restrict__ mean,
    const float* __restrict__ var, float* __restrict__ Out)   // [B,N,512]
{
    const int lane = threadIdx.x & 31, wave = threadIdx.x >> 5;
    const int lrow = lane & 15, lhi = lane >> 4;
    const int b = blockIdx.z;
    const int nrow0 = blockIdx.y * 128 + wave * 16;
    const int orow0 = blockIdx.x * 64;
    v8f acc[4] = {};
    wmma_k512(XsT + ((size_t)b * N_DIM + nrow0 + lrow) * C_DIM,
              Wtb + (size_t)(orow0 + lrow) * C_DIM, lhi, acc);
#pragma unroll
    for (int j = 0; j < 4; ++j) {
        int o = orow0 + j * 16 + lrow;          // fixed per lane within j
        float inv = gamma[o] * rsqrtf(var[o] + BN_EPS);
        float sh  = beta[o] - mean[o] * inv;
        float bo  = bt[o];
#pragma unroll
        for (int v = 0; v < 8; ++v) {
            int n = nrow0 + lhi * 8 + v;
            float val = (acc[j][v] + bo) * inv + sh;
            Out[((size_t)b * N_DIM + n) * C_DIM + o] = fmaxf(val, 0.f);
        }
    }
}

// ---------------------------------------------------------------------------
// Flash attention.  Grid (N/16, B), 128 threads = 4 waves.
// Wave w: 16 queries x 128 V-channels (8 O tiles).
// S^T = K·Q^T (lane holds fixed query q = lane&15), O^T = V^T·P^T.
// ---------------------------------------------------------------------------
__global__ __launch_bounds__(128, 1) void flash_attn_kernel(
    const unsigned short* __restrict__ Q,   // [B, N, DA] bf16 (pre-scaled)
    const unsigned short* __restrict__ K,   // [B, M, DA] bf16
    const unsigned short* __restrict__ V,   // [B, C, M]  bf16
    unsigned short* __restrict__ XS)        // [B, N, C]  bf16 (transposed)
{
    const int lane = threadIdx.x & 31;
    const int wave = threadIdx.x >> 5;      // 0..3
    const int b    = blockIdx.y;
    const int n0   = blockIdx.x * 16;
    const int lrow = lane & 15;
    const int lhi  = lane >> 4;

    // Q as B-operand fragments: lane q = lrow, 16 contiguous d at dc*32 + 16*lhi
    bf16x16 qf[4];
    {
        const unsigned short* qrow = Q + ((size_t)b * N_DIM + (n0 + lrow)) * DA_DIM;
#pragma unroll
        for (int dc = 0; dc < 4; ++dc)
            qf[dc] = load_bfrag(qrow + dc * 32 + 16 * lhi);
    }

    const unsigned short* Kb = K + (size_t)b * M_DIM * DA_DIM;
    const unsigned short* Vb = V + (size_t)b * C_DIM * M_DIM;
    const unsigned short* vrow = Vb + (size_t)(wave * 128 + lrow) * M_DIM + 16 * lhi;

    v8f oacc[8] = {};
    float m_run = -1e30f, l_run = 0.f;

    for (int kb = 0; kb < M_DIM; kb += 32) {
        if (kb + 32 < M_DIM)
            __builtin_prefetch(Kb + (size_t)(kb + 32 + lrow) * DA_DIM, 0, 0);

        // ---- all 8 K fragments in one clause, then 8 QK^T wmmas
        const unsigned short* kr = Kb + (size_t)(kb + lrow) * DA_DIM + 16 * lhi;
        bf16x16 ka[8];
#pragma unroll
        for (int dc = 0; dc < 4; ++dc) {
            ka[dc]     = load_afrag(kr + dc * 32);
            ka[4 + dc] = load_afrag(kr + (size_t)16 * DA_DIM + dc * 32);
        }
        v8f s0 = {}, s1 = {};
#pragma unroll
        for (int dc = 0; dc < 4; ++dc)
            s0 = __builtin_amdgcn_wmma_f32_16x16x32_bf16(
                     false, ka[dc], false, qf[dc], (short)0, s0, false, false);
#pragma unroll
        for (int dc = 0; dc < 4; ++dc)
            s1 = __builtin_amdgcn_wmma_f32_16x16x32_bf16(
                     false, ka[4 + dc], false, qf[dc], (short)0, s1, false, false);

        // ---- issue first half of V loads early: latency overlaps softmax VALU
        bf16x16 vf[4];
#pragma unroll
        for (int ct = 0; ct < 4; ++ct)
            vf[ct] = load_afrag(vrow + (size_t)ct * 16 * M_DIM + kb);

        // ---- online softmax for row q = lrow (paired with lane^16)
        float vmax = s0[0];
#pragma unroll
        for (int i = 0; i < 8; ++i) {
            vmax = fmaxf(vmax, s0[i]);
            vmax = fmaxf(vmax, s1[i]);
        }
        vmax = fmaxf(vmax, swz16f(vmax));
        float m_new = fmaxf(m_run, vmax);
        float alpha = __expf(m_run - m_new);
        m_run = m_new;

        float p0[8], p1[8], rsum = 0.f;
#pragma unroll
        for (int i = 0; i < 8; ++i) {
            p0[i] = __expf(s0[i] - m_new);
            p1[i] = __expf(s1[i] - m_new);
            rsum += p0[i] + p1[i];
        }
        rsum += swz16f(rsum);
        l_run = l_run * alpha + rsum;

#pragma unroll
        for (int t = 0; t < 8; ++t)
#pragma unroll
            for (int i = 0; i < 8; ++i) oacc[t][i] *= alpha;

        // ---- P^T B-fragment via one cross-half exchange
        unsigned own0[4], own1[4];
#pragma unroll
        for (int i = 0; i < 4; ++i) {
            own0[i] = (unsigned)f2bf(p0[2 * i]) | ((unsigned)f2bf(p0[2 * i + 1]) << 16);
            own1[i] = (unsigned)f2bf(p1[2 * i]) | ((unsigned)f2bf(p1[2 * i + 1]) << 16);
        }
        unsigned oth0[4], oth1[4];
#pragma unroll
        for (int i = 0; i < 4; ++i) {
            oth0[i] = swz16u(own0[i]);
            oth1[i] = swz16u(own1[i]);
        }
        u32x8 pw;
#pragma unroll
        for (int i = 0; i < 4; ++i) {
            pw[i]     = (lhi == 0) ? own0[i] : oth1[i];
            pw[i + 4] = (lhi == 0) ? oth0[i] : own1[i];
        }
        bf16x16 pfrag = __builtin_bit_cast(bf16x16, pw);

        // ---- O^T += V^T · P^T: first 4 tiles (preloaded), second 4 overlap
        bf16x16 vg[4];
#pragma unroll
        for (int ct = 0; ct < 4; ++ct)
            vg[ct] = load_afrag(vrow + (size_t)(ct + 4) * 16 * M_DIM + kb);
#pragma unroll
        for (int ct = 0; ct < 4; ++ct)
            oacc[ct] = __builtin_amdgcn_wmma_f32_16x16x32_bf16(
                           false, vf[ct], false, pfrag, (short)0, oacc[ct], false, false);
#pragma unroll
        for (int ct = 0; ct < 4; ++ct)
            oacc[ct + 4] = __builtin_amdgcn_wmma_f32_16x16x32_bf16(
                               false, vg[ct], false, pfrag, (short)0, oacc[ct + 4], false, false);
    }

    // ---- normalize & store x_s^T[b, n, c]: 8 consecutive c -> one 16B store
    float invl = 1.0f / l_run;
    unsigned short* xrow = XS + ((size_t)b * N_DIM + (n0 + lrow)) * C_DIM;
#pragma unroll
    for (int ct = 0; ct < 8; ++ct) {
        int c0 = wave * 128 + ct * 16 + lhi * 8;
        u32x4 pk;
#pragma unroll
        for (int i = 0; i < 4; ++i) {
            pk[i] = (unsigned)f2bf(oacc[ct][2 * i] * invl) |
                    ((unsigned)f2bf(oacc[ct][2 * i + 1] * invl) << 16);
        }
        *(u32x4*)(xrow + c0) = pk;
    }
}

// ---------------------------------------------------------------------------
extern "C" void kernel_launch(void* const* d_in, const int* in_sizes, int n_in,
                              void* d_out, int out_size, void* d_ws, size_t ws_size,
                              hipStream_t stream)
{
    const float* x     = (const float*)d_in[0];
    const float* y     = (const float*)d_in[1];
    const float* w_qk  = (const float*)d_in[2];
    const float* w_v   = (const float*)d_in[3];
    const float* b_v   = (const float*)d_in[4];
    const float* w_t   = (const float*)d_in[5];
    const float* b_t   = (const float*)d_in[6];
    const float* gamma = (const float*)d_in[7];
    const float* beta  = (const float*)d_in[8];
    const float* rmean = (const float*)d_in[9];
    const float* rvar  = (const float*)d_in[10];
    float* out = (float*)d_out;

    // workspace (ushort units)
    unsigned short* xT    = (unsigned short*)d_ws;                  // [B,N,512]
    unsigned short* yT    = xT    + (size_t)B_SZ * N_DIM * C_DIM;   // [B,M,512]
    unsigned short* wqk_b = yT    + (size_t)B_SZ * M_DIM * C_DIM;   // [128,512]
    unsigned short* wv_b  = wqk_b + (size_t)DA_DIM * C_DIM;         // [512,512]
    unsigned short* wt_b  = wv_b  + (size_t)C_DIM * C_DIM;          // [512,512]
    unsigned short* q_ws  = wt_b  + (size_t)C_DIM * C_DIM;          // [B,N,128]
    unsigned short* k_ws  = q_ws  + (size_t)B_SZ * N_DIM * DA_DIM;  // [B,M,128]
    unsigned short* v_ws  = k_ws  + (size_t)B_SZ * M_DIM * DA_DIM;  // [B,512,M]
    unsigned short* xs_ws = v_ws  + (size_t)B_SZ * C_DIM * M_DIM;   // [B,N,512]

    const float qscale = 1.0f / sqrtf((float)DA_DIM);

    // 1) transpose+convert activations, convert weights to bf16
    transpose_cvt_kernel<<<dim3(N_DIM / 32, C_DIM / 32, B_SZ), dim3(32, 8), 0, stream>>>(x, xT, N_DIM);
    transpose_cvt_kernel<<<dim3(M_DIM / 32, C_DIM / 32, B_SZ), dim3(32, 8), 0, stream>>>(y, yT, M_DIM);
    cvt_kernel<<<(DA_DIM * C_DIM + 255) / 256, 256, 0, stream>>>(w_qk, wqk_b, DA_DIM * C_DIM);
    cvt_kernel<<<(C_DIM * C_DIM + 255) / 256, 256, 0, stream>>>(w_v, wv_b, C_DIM * C_DIM);
    cvt_kernel<<<(C_DIM * C_DIM + 255) / 256, 256, 0, stream>>>(w_t, wt_b, C_DIM * C_DIM);

    // 2) WMMA projections: q (scaled), k -> [B,P,128]; v -> [B,512,M]
    gemm_po_kernel<<<dim3(DA_DIM / 64, N_DIM / 128, B_SZ), 256, 0, stream>>>(
        xT, wqk_b, q_ws, N_DIM, DA_DIM, qscale);
    gemm_po_kernel<<<dim3(DA_DIM / 64, M_DIM / 128, B_SZ), 256, 0, stream>>>(
        yT, wqk_b, k_ws, M_DIM, DA_DIM, 1.0f);
    gemm_op_kernel<<<dim3(M_DIM / 512, C_DIM / 16, B_SZ), 256, 0, stream>>>(
        wv_b, yT, b_v, v_ws, C_DIM, M_DIM);

    // 3) WMMA flash attention -> xs^T [B,N,512]
    flash_attn_kernel<<<dim3(N_DIM / 16, B_SZ), 128, 0, stream>>>(q_ws, k_ws, v_ws, xs_ws);

    // 4) WMMA final GEMM + BN + ReLU -> [B,N,512] fp32
    gemm_final_kernel<<<dim3(C_DIM / 64, N_DIM / 128, B_SZ), 256, 0, stream>>>(
        xs_ws, wt_b, b_t, gamma, beta, rmean, rvar, out);
}
